// HipSelfAttention_31430570672227
// MI455X (gfx1250) — matrix operand order repaired and checked by reading, not compiled, not run on hardware
//
#include <hip/hip_runtime.h>
#include <hip/hip_bf16.h>

typedef __bf16 bf16_t;
typedef __attribute__((ext_vector_type(16))) __bf16 v16bf;
typedef __attribute__((ext_vector_type(8)))  __bf16 v8bf;
typedef __attribute__((ext_vector_type(8)))  float  v8f;

#define B_ 2
#define S_ 4096
#define H_ 16
#define D_ 64
#define KT 32              // keys per tile
#define NTILES (S_ / KT)   // 128

__device__ __forceinline__ v8f wmma_bf16(v16bf a, v16bf b, v8f c) {
  // D = A(16x32 bf16) * B(32x16 bf16) + C(16x16 f32)
  return __builtin_amdgcn_wmma_f32_16x16x32_bf16(
      /*neg_a=*/false, a, /*neg_b=*/false, b,
      /*c_mod=*/(short)0, c, /*reuse_a=*/false, /*reuse_b=*/false);
}

// gfx1250 async copy: global -> LDS, 16 bytes per lane, tracked by ASYNCcnt.
__device__ __forceinline__ void async_copy_b128(uint32_t lds_byte_addr, uint64_t gaddr) {
  asm volatile("global_load_async_to_lds_b128 %0, %1, off"
               :: "v"(lds_byte_addr), "v"(gaddr)
               : "memory");
}
__device__ __forceinline__ void wait_async_le2() {
  asm volatile("s_wait_asynccnt 0x2" ::: "memory");
}
__device__ __forceinline__ void wait_async_le0() {
  asm volatile("s_wait_asynccnt 0x0" ::: "memory");
}
__device__ __forceinline__ uint32_t lds_addr_of(const void* p) {
  // Generic pointers into LDS carry the LDS byte offset in the low 32 bits.
  return (uint32_t)(uintptr_t)p;
}

// qkv: [B, S, 3, H, D] bf16 ; out: [B, S, H, D] bf16
__global__ __launch_bounds__(256) void fa_fwd_bf16(const bf16_t* __restrict__ qkv,
                                                   bf16_t* __restrict__ out) {
  __shared__ bf16_t lk[2][KT * D_];   // K tiles, double buffered (4KB each)
  __shared__ bf16_t lv[2][KT * D_];   // V tiles, double buffered (4KB each)

  const int lane = threadIdx.x & 31;
  const int wave = threadIdx.x >> 5;
  const int l16  = lane & 15;
  const int half = lane >> 4;

  const int qtile = blockIdx.x;        // 0..S/128-1
  const int bh    = blockIdx.y;        // 0..B*H-1
  const int b     = bh >> 4;           // /H_
  const int h     = bh & (H_ - 1);
  const int q0    = qtile * 128 + wave * 16;   // first query row of this wave

  const size_t sstride = (size_t)3 * H_ * D_;  // elements per sequence position
  const size_t base    = (size_t)b * S_ * sstride + (size_t)h * D_;
  const bf16_t* qp = qkv + base;                        // which = 0
  const bf16_t* kp = qkv + base + (size_t)H_ * D_;      // which = 1
  const bf16_t* vp = qkv + base + (size_t)2 * H_ * D_;  // which = 2

  // Cooperative copy mapping: 8 lanes per 128B row, 256 threads per 32-row tile.
  const int cr = threadIdx.x >> 3;        // 0..31  (key row in tile)
  const int cc = (threadIdx.x & 7) * 8;   // element offset (16B chunks)

  // ---- Q^T B-operands (d x query), steps s=0,1 covering d=0..31 / 32..63.
  // B layout (32x16 bf16): lane holds column n = query l16; elems j -> d = 8*half + j (lo), 16 + 8*half + j (hi).
  // Fold (1/sqrt(D)) * log2(e) into Q so scores are natively in log2 units:
  // softmax then needs only subtract + v_exp_f32 per element.
  const float qscale = 0.125f * 1.44269504088896340736f;
  v16bf qt[2];
  {
    const int qrow = q0 + l16;
#pragma unroll
    for (int s = 0; s < 2; ++s) {
#pragma unroll
      for (int j = 0; j < 16; ++j) {
        const int d = s * 32 + (j < 8 ? 8 * half + j : 16 + 8 * half + (j - 8));
        qt[s][j] = (bf16_t)((float)qp[(size_t)qrow * sstride + d] * qscale);
      }
    }
  }

  // Per-query running stats (query q = l16, replicated across both half-lanes)
  float m_run = -1e30f;   // in log2 units
  float l_run = 0.0f;
  v8f   o[4] = {v8f{}, v8f{}, v8f{}, v8f{}};   // O accum: 16q x 64d as 4 C-tiles

  // ---- Prologue: async-stage tile 0 into buffer 0.
  {
    const size_t rowoff = (size_t)cr * sstride + cc;
    async_copy_b128(lds_addr_of(&lk[0][cr * D_ + cc]), (uint64_t)(uintptr_t)(kp + rowoff));
    async_copy_b128(lds_addr_of(&lv[0][cr * D_ + cc]), (uint64_t)(uintptr_t)(vp + rowoff));
  }

  for (int it = 0; it < NTILES; ++it) {
    const int buf = it & 1;

    // Issue next tile's async copies into the other buffer (safe: the last
    // compute touching that buffer finished before the previous barrier).
    if (it + 1 < NTILES) {
      const size_t rowoff = (size_t)(it + 1) * KT * sstride + (size_t)cr * sstride + cc;
      async_copy_b128(lds_addr_of(&lk[buf ^ 1][cr * D_ + cc]),
                      (uint64_t)(uintptr_t)(kp + rowoff));
      async_copy_b128(lds_addr_of(&lv[buf ^ 1][cr * D_ + cc]),
                      (uint64_t)(uintptr_t)(vp + rowoff));
      wait_async_le2();   // async loads complete in order -> tile `it` resident
    } else {
      wait_async_le0();
    }
    __syncthreads();      // all waves' portions of tile `it` are in LDS

    // ---- All 16 fragment loads issued as one batch (see sched_barrier below)
    // K as A-operands: tiles t=0,1 (16 keys), k-steps s=0,1.
    // A layout (16x32 bf16): lane row = l16; elems j -> K = 8*half + j (lo),
    // 16 + 8*half + j (hi).
    v16bf ka[2][2];
    {
      const int dbase = 8 * half;
#pragma unroll
      for (int t = 0; t < 2; ++t) {
        const bf16_t* krow = &lk[buf][(t * 16 + l16) * D_];
#pragma unroll
        for (int s = 0; s < 2; ++s) {
          v8bf lo = *(const v8bf*)(krow + s * 32 + dbase);
          v8bf hi = *(const v8bf*)(krow + s * 32 + 16 + dbase);
#pragma unroll
          for (int j = 0; j < 8; ++j) { ka[t][s][j] = lo[j]; ka[t][s][8 + j] = hi[j]; }
        }
      }
    }
    // V as B-operands: nb = 0..3 (d blocks of 16). Lane holds d column
    // nb*16 + l16; elems j -> key 8*half + j (lo), 16 + 8*half + j (hi).
    v16bf vb[4];
    {
      const bf16_t* vcol = &lv[buf][l16];
#pragma unroll
      for (int nb = 0; nb < 4; ++nb) {
#pragma unroll
        for (int j = 0; j < 8; ++j) {
          vb[nb][j]     = vcol[(8 * half + j) * D_ + nb * 16];
          vb[nb][8 + j] = vcol[(16 + 8 * half + j) * D_ + nb * 16];
        }
      }
    }
    // Keep the ds_load clause ahead of the math: pay LDS latency once.
    __builtin_amdgcn_sched_barrier(0);

    // ---- Scores transposed: S^T = K * Q^T  (two 16key x 16query tiles),
    // already in log2 units thanks to the pre-scaled Q.
    v8f st0 = v8f{}, st1 = v8f{};
    st0 = wmma_bf16(ka[0][0], qt[0], st0);
    st0 = wmma_bf16(ka[0][1], qt[1], st0);
    st1 = wmma_bf16(ka[1][0], qt[0], st1);
    st1 = wmma_bf16(ka[1][1], qt[1], st1);

    // ---- Online softmax. C-layout of st: lane -> query l16, VGPR v -> key
    // v + 8*half (tile0) / 16 + v + 8*half (tile1).
    float mx = -1e30f;
#pragma unroll
    for (int v = 0; v < 8; ++v) {
      mx = fmaxf(mx, st0[v]);
      mx = fmaxf(mx, st1[v]);
    }
    mx = fmaxf(mx, __shfl_xor(mx, 16, 32));        // combine half-lanes
    const float m_new = fmaxf(m_run, mx);
    const float alpha = __builtin_amdgcn_exp2f(m_run - m_new);

    // P = exp2(S - m_new), packed straight into the A-operand layout for the
    // O GEMM (no cross-lane movement needed).
    v16bf pa;
    float rs = 0.0f;
#pragma unroll
    for (int v = 0; v < 8; ++v) {
      float p0 = __builtin_amdgcn_exp2f(st0[v] - m_new);
      float p1 = __builtin_amdgcn_exp2f(st1[v] - m_new);
      rs += p0 + p1;
      pa[v]     = (bf16_t)p0;   // keys  8*half + v
      pa[8 + v] = (bf16_t)p1;   // keys 16 + 8*half + v
    }
    rs += __shfl_xor(rs, 16, 32);
    l_run = l_run * alpha + rs;
    m_run = m_new;

    // Rescale O accumulators; O C-layout: lane -> d (within nb), VGPR v ->
    // query v + 8*half, so fetch alpha for that query from stats lanes 0..15.
#pragma unroll
    for (int v = 0; v < 8; ++v) {
      const float av = __shfl(alpha, v + 8 * half, 32);
      o[0][v] *= av; o[1][v] *= av; o[2][v] *= av; o[3][v] *= av;
    }

    // ---- O += P * V
#pragma unroll
    for (int nb = 0; nb < 4; ++nb)
      o[nb] = wmma_bf16(pa, vb[nb], o[nb]);

    __syncthreads();      // tile `it` consumed; its buffer may be re-filled
  }

  // ---- Epilogue: normalize by row-sum and store bf16.
  const float linv = 1.0f / l_run;
#pragma unroll
  for (int v = 0; v < 8; ++v) {
    const float lv = __shfl(linv, v + 8 * half, 32);
    const int q = q0 + v + 8 * half;
    const size_t orow = ((size_t)(b * S_ + q) * H_ + h) * D_;
#pragma unroll
    for (int nb = 0; nb < 4; ++nb)
      out[orow + nb * 16 + l16] = (bf16_t)(o[nb][v] * lv);
  }
}

extern "C" void kernel_launch(void* const* d_in, const int* in_sizes, int n_in,
                              void* d_out, int out_size, void* d_ws, size_t ws_size,
                              hipStream_t stream) {
  (void)in_sizes; (void)n_in; (void)out_size; (void)d_ws; (void)ws_size;
  const bf16_t* qkv = (const bf16_t*)d_in[0];
  bf16_t* out = (bf16_t*)d_out;
  dim3 grid(S_ / 128, B_ * H_);   // 32 query tiles x 32 (b,h) pairs
  fa_fwd_bf16<<<grid, 256, 0, stream>>>(qkv, out);
}
